// AttentionHead_33449205301259
// MI455X (gfx1250) — matrix operand order repaired and checked by reading, not compiled, not run on hardware
//
#include <hip/hip_runtime.h>

// ---------------- CDNA5 WMMA types ----------------
typedef __attribute__((ext_vector_type(16))) __bf16 v16bf;
typedef __attribute__((ext_vector_type(8)))  __bf16 v8bf;
typedef __attribute__((ext_vector_type(8)))  float  v8f;
typedef __attribute__((ext_vector_type(4)))  unsigned int v4u;
typedef __attribute__((ext_vector_type(8)))  int v8i;
typedef __attribute__((ext_vector_type(4)))  int v4i;

#define WMMA_BF16(a, b, c) \
    __builtin_amdgcn_wmma_f32_16x16x32_bf16(false, (a), false, (b), (short)0, (c), false, false)

// B=8, T=2048, E=1024, H=64
// ws layout (bytes):
//   Wt  bf16 [192][1024]        @ 0        (384 KB)  -- [Wq|Wk|Wv] transposed, bf16
//   Qb  bf16 [B*T][64]          @ 393216   (2 MB)    -- pre-scaled by 1/8
//   Kb  bf16 [B*T][64]          @ 2490368  (2 MB)
//   Vt  bf16 [B][64][2048]      @ 4587520  (2 MB)    -- V transposed (h-major)

// ---------------- kernel 0: weight pack (transpose + fp32->bf16) ----------------
__global__ __launch_bounds__(256) void pack_weights(const float* __restrict__ Wq,
                                                    const float* __restrict__ Wk,
                                                    const float* __restrict__ Wv,
                                                    __bf16* __restrict__ Wt) {
    int idx = blockIdx.x * 256 + threadIdx.x;      // 192*1024 elements
    if (idx >= 192 * 1024) return;
    int row = idx >> 10;                            // 0..191  (s*64 + n)
    int k   = idx & 1023;
    int s = row >> 6, n = row & 63;
    const float* W = (s == 0) ? Wq : (s == 1) ? Wk : Wv;
    Wt[idx] = (__bf16)W[k * 64 + n];
}

// ---------------- TDM: async 2D tile load  Wt[192 rows][32 k] -> LDS ----------------
// data_size=2B, tile_dim0=32 (64B rows), tile_dim1=192, tensor stride 1024 elems.
// pad_enable: +4 DWORDs (16B) after every 16 DWORDs (64B) -> LDS row stride 80B (40 bf16).
__device__ __forceinline__ void tdm_load_w(unsigned lds_off, const __bf16* gptr) {
    unsigned long long ga = (unsigned long long)(uintptr_t)gptr;
    v4u g0;
    g0[0] = 1u;                                              // count=1, user descriptor
    g0[1] = lds_off;                                         // lds_addr (bytes)
    g0[2] = (unsigned)(ga & 0xffffffffu);                    // global_addr lo
    g0[3] = (unsigned)((ga >> 32) & 0x01ffffffu) | (2u << 30); // global_addr hi | type=2
    v8i g1;
    g1[0] = (1 << 16)        // data_size = 2 bytes
          | (1 << 20)        // pad_enable
          | (3 << 22)        // pad_interval: 16 DWORDs (64B)
          | (3 << 25);       // pad_amount:   4 DWORDs (16B)
    g1[1] = (int)(1024u << 16);   // tensor_dim0 = 1024 (lo16 @ bits 63:48)
    g1[2] = (int)(192u  << 16);   // tensor_dim0 hi = 0 | tensor_dim1 = 192 (lo16)
    g1[3] = (int)(32u   << 16);   // tensor_dim1 hi = 0 | tile_dim0 = 32
    g1[4] = 192;                  // tile_dim1 = 192, tile_dim2 = 0
    g1[5] = 1024;                 // tensor_dim0_stride lo32
    g1[6] = 0;                    // stride hi | tensor_dim1_stride lo
    g1[7] = 0;
    v4i z4 = {};
    v8i z8 = {};
    // amdgpu-toolchain (clang-23) 6-arg form: groups 0..3 + extra v8i + cpol
    __builtin_amdgcn_tensor_load_to_lds(g0, g1, z4, z4, z8, 0);
}

// ---------------- kernel 1: fused QKV projection (bf16 WMMA, fp32 accum) ----------------
// Block = 4 waves, 64 rows. W chunk staged via TDM into double-buffered LDS one
// k-step ahead; all waves share it. WMMAs only wait on short ds_loads.
__global__ __launch_bounds__(128) void qkv_proj(const float*  __restrict__ x,
                                                const __bf16* __restrict__ Wt,
                                                __bf16* __restrict__ Qb,
                                                __bf16* __restrict__ Kb,
                                                __bf16* __restrict__ Vt) {
    __shared__ __bf16 Wl[2][192 * 40];   // 2 x 15360B, row stride 40 bf16 (80B)

    const int lane = threadIdx.x & 31;
    const int wave = threadIdx.x >> 5;
    const int l16  = lane & 15;
    const int half = lane >> 4;
    const int rb   = (blockIdx.x * 4 + wave) * 16;  // global row base (of 16384)

    v8f acc[12];
#pragma unroll
    for (int i = 0; i < 12; i++) acc[i] = (v8f){};

    const float* xl = x + (size_t)(rb + l16) * 1024 + half * 16;

    // prologue: A load (k=0) + TDM of W chunk 0
    v8f lo = *(const v8f*)(xl);
    v8f hi = *(const v8f*)(xl + 8);
    if (wave == 0)
        tdm_load_w((unsigned)(uintptr_t)&Wl[0][0], Wt);

    for (int ks = 0; ks < 32; ks++) {
        const int k = ks * 32;
        if (wave == 0) {
            if (ks + 1 < 32) {   // issue next chunk's DMA, then wait for current (older)
                tdm_load_w((unsigned)(uintptr_t)&Wl[(ks + 1) & 1][0], Wt + k + 32);
                __builtin_amdgcn_s_wait_tensorcnt(1);
            } else {
                __builtin_amdgcn_s_wait_tensorcnt(0);
            }
        }
        __syncthreads();         // publish Wl[ks&1]

        v16bf a;
#pragma unroll
        for (int i = 0; i < 8; i++) { a[i] = (__bf16)lo[i]; a[i + 8] = (__bf16)hi[i]; }
        if (k + 32 < 1024) {     // prefetch next A chunk under this step's WMMAs
            lo = *(const v8f*)(xl + k + 32);
            hi = *(const v8f*)(xl + k + 40);
        }

        const __bf16* wb = &Wl[ks & 1][0] + (size_t)l16 * 40 + half * 16;
#pragma unroll
        for (int nt = 0; nt < 12; nt++) {
            v16bf b = *(const v16bf*)(wb + nt * 16 * 40);   // 32B conflict-free LDS read
            acc[nt] = WMMA_BF16(a, b, acc[nt]);
        }
        __syncthreads();         // all waves done reading before buffer is overwritten
    }

    const int batch = rb >> 11;       // rb / 2048
    const int t0    = rb & 2047;
#pragma unroll
    for (int nt = 0; nt < 4; nt++) {
#pragma unroll
        for (int r = 0; r < 8; r++) {
            const int row = rb + r + 8 * half;
            Qb[(size_t)row * 64 + nt * 16 + l16] = (__bf16)(acc[nt][r] * 0.125f);   // * H^-1/2
            Kb[(size_t)row * 64 + nt * 16 + l16] = (__bf16)(acc[nt + 4][r]);
        }
        // V transposed store: consecutive t per lane -> one 16B packed store
        const int h = nt * 16 + l16;
        v8bf vv;
#pragma unroll
        for (int r = 0; r < 8; r++) vv[r] = (__bf16)(acc[nt + 8][r]);
        *(v8bf*)(Vt + ((size_t)batch * 64 + h) * 2048 + t0 + 8 * half) = vv;
    }
}

// ---------------- wave32 16-lane reductions via ds_swizzle (xor within half) ----------------
template <int XOR>
__device__ __forceinline__ float swz(float v) {
    return __int_as_float(__builtin_amdgcn_ds_swizzle(__float_as_int(v), (XOR << 10) | 0x1f));
}
__device__ __forceinline__ float rmax16(float v) {
    v = fmaxf(v, swz<1>(v)); v = fmaxf(v, swz<2>(v));
    v = fmaxf(v, swz<4>(v)); v = fmaxf(v, swz<8>(v));
    return v;
}
__device__ __forceinline__ float rsum16(float v) {
    v += swz<1>(v); v += swz<2>(v); v += swz<4>(v); v += swz<8>(v);
    return v;
}

// ---------------- kernel 2: causal flash attention ----------------
// Each wave: 16 query rows; loops over 32-wide key blocks up to its causal bound.
// All K/V B-operand loads issued at loop top so V latency hides under softmax VALU work.
__global__ __launch_bounds__(128) void flash_attn(const __bf16* __restrict__ Qb,
                                                  const __bf16* __restrict__ Kb,
                                                  const __bf16* __restrict__ Vt,
                                                  float* __restrict__ out) {
    __shared__ __bf16 Pl[4][16][40];   // per-wave P staging, padded rows (80B stride)

    const int lane = threadIdx.x & 31;
    const int wave = threadIdx.x >> 5;
    const int l16  = lane & 15;
    const int half = lane >> 4;
    const int batch = blockIdx.y;
    const int qw = (blockIdx.x * 4 + wave) * 16;           // query row base in batch

    // Q A-operands (resident across whole key loop)
    const __bf16* qrow = Qb + ((size_t)batch * 2048 + qw + l16) * 64;
    v16bf qa[2];
#pragma unroll
    for (int c = 0; c < 2; c++) {
        v8bf qlo = *(const v8bf*)(qrow + 32 * c + half * 16);
        v8bf qhi = *(const v8bf*)(qrow + 32 * c + half * 16 + 8);
#pragma unroll
        for (int i = 0; i < 8; i++) { qa[c][i] = qlo[i]; qa[c][8 + i] = qhi[i]; }
    }

    float m[8], l[8];
    v8f o[4];
#pragma unroll
    for (int r = 0; r < 8; r++) { m[r] = -1e30f; l[r] = 0.f; }
#pragma unroll
    for (int nt = 0; nt < 4; nt++) o[nt] = (v8f){};

    const __bf16* kbase = Kb + (size_t)batch * 2048 * 64;
    const __bf16* vbase = Vt + (size_t)batch * 64 * 2048;
    const int nkb = (qw + 47) >> 5;                         // causal block bound

    for (int kb = 0; kb < nkb; kb++) {
        const int kb32 = kb * 32;

        // ---- issue ALL loads for this block up front ----
        v16bf kop[2][2];
#pragma unroll
        for (int jt = 0; jt < 2; jt++) {
            const __bf16* krow = kbase + (size_t)(kb32 + jt * 16 + l16) * 64;
            kop[jt][0] = *(const v16bf*)(krow + half * 16);
            kop[jt][1] = *(const v16bf*)(krow + 32 + half * 16);
        }
        v16bf bv[4];
#pragma unroll
        for (int nt = 0; nt < 4; nt++)
            bv[nt] = *(const v16bf*)(vbase + (size_t)(nt * 16 + l16) * 2048 + kb32 + half * 16);
        // prefetch next key block (global_prefetch_b8)
        if (kb + 1 < nkb) {
            __builtin_prefetch(kbase + (size_t)(kb32 + 32 + l16) * 64, 0, 1);
            __builtin_prefetch(vbase + (size_t)l16 * 2048 + kb32 + 32, 0, 1);
        }

        // ---- S = Q K^T (4 WMMAs) ----
        v8f s[2];
#pragma unroll
        for (int jt = 0; jt < 2; jt++) {
            v8f c = (v8f){};
            c = WMMA_BF16(qa[0], kop[jt][0], c);
            c = WMMA_BF16(qa[1], kop[jt][1], c);
            s[jt] = c;
        }
        // ---- causal mask (only partial blocks) ----
        if (kb32 + 31 > qw) {
#pragma unroll
            for (int jt = 0; jt < 2; jt++) {
                const int col = kb32 + jt * 16 + l16;
#pragma unroll
                for (int r = 0; r < 8; r++) {
                    const int row = qw + r + 8 * half;
                    if (col > row) s[jt][r] = -1e30f;
                }
            }
        }
        // ---- online softmax update (V loads completing in background) ----
#pragma unroll
        for (int r = 0; r < 8; r++) {
            float mx = fmaxf(rmax16(s[0][r]), rmax16(s[1][r]));
            float mn = fmaxf(m[r], mx);
            float corr = __expf(m[r] - mn);
            float p0 = __expf(s[0][r] - mn);
            float p1 = __expf(s[1][r] - mn);
            s[0][r] = p0; s[1][r] = p1;
            l[r] = l[r] * corr + (rsum16(p0) + rsum16(p1));
            m[r] = mn;
#pragma unroll
            for (int nt = 0; nt < 4; nt++) o[nt][r] *= corr;
        }
        // ---- C-layout -> A-layout for P via per-wave LDS slice ----
#pragma unroll
        for (int jt = 0; jt < 2; jt++)
#pragma unroll
            for (int r = 0; r < 8; r++)
                Pl[wave][r + 8 * half][jt * 16 + l16] = (__bf16)s[jt][r];
        // same-wave LDS ordering: drain DS pipe, block compiler reordering
        asm volatile("s_wait_dscnt 0" ::: "memory");
        v8bf plo = *(const v8bf*)&Pl[wave][l16][half * 16];
        v8bf phi = *(const v8bf*)&Pl[wave][l16][half * 16 + 8];
        v16bf pa;
#pragma unroll
        for (int i = 0; i < 8; i++) { pa[i] = plo[i]; pa[8 + i] = phi[i]; }
        // ---- O += P V (4 WMMAs; V operands already resident) ----
#pragma unroll
        for (int nt = 0; nt < 4; nt++)
            o[nt] = WMMA_BF16(pa, bv[nt], o[nt]);
    }

    // ---- epilogue: O / l -> out [B,T,64] fp32 ----
#pragma unroll
    for (int r = 0; r < 8; r++) {
        const float inv = 1.0f / l[r];
        const int row = qw + r + 8 * half;
        float* orow = out + ((size_t)batch * 2048 + row) * 64;
#pragma unroll
        for (int nt = 0; nt < 4; nt++)
            orow[nt * 16 + l16] = o[nt][r] * inv;
    }
}

// ---------------- host launcher ----------------
extern "C" void kernel_launch(void* const* d_in, const int* in_sizes, int n_in,
                              void* d_out, int out_size, void* d_ws, size_t ws_size,
                              hipStream_t stream) {
    const float* x  = (const float*)d_in[0];
    const float* Wq = (const float*)d_in[1];
    const float* Wk = (const float*)d_in[2];
    const float* Wv = (const float*)d_in[3];

    char* ws = (char*)d_ws;
    __bf16* Wt = (__bf16*)(ws);
    __bf16* Qb = (__bf16*)(ws + 393216);
    __bf16* Kb = (__bf16*)(ws + 393216 + 2097152);
    __bf16* Vt = (__bf16*)(ws + 393216 + 2 * 2097152);
    float* out = (float*)d_out;

    pack_weights<<<768, 256, 0, stream>>>(Wq, Wk, Wv, Wt);      // 192*1024 elems
    qkv_proj<<<256, 128, 0, stream>>>(x, Wt, Qb, Kb, Vt);       // 16384 rows / (4 waves * 16)
    dim3 grid(32, 8);                                            // 2048/(4*16) tiles x 8 batches
    flash_attn<<<grid, 128, 0, stream>>>(Qb, Kb, Vt, out);
}
